// PINN_6828998000998
// MI455X (gfx1250) — matrix operand order repaired and checked by reading, not compiled, not run on hardware
//
#include <hip/hip_runtime.h>
#include <math.h>

// ---------------- CDNA5 WMMA types ----------------
typedef __attribute__((ext_vector_type(16))) __bf16 v16bf;
typedef __attribute__((ext_vector_type(8)))  __bf16 v8bf;
typedef __attribute__((ext_vector_type(8)))  float  v8f;

// ---------------- problem constants ----------------
#define N_PTS  32768
#define N_QUAD 1000
#define N_DATA 1024
#define NPTOT  (N_PTS + N_QUAD + N_DATA + 2)   // 34794 evaluation points
#define NPPAD  34816                           // padded to multiple of 16
#define NTILES (NPPAD / 16)                    // 2176 tiles of 16 points
#define HS     136                             // padded LDS row stride (bf16)
#define W_CONST 1.5f                           // W_C = X_C/EI = 1.5 exactly

// d_ws byte offsets (total ~250 KB)
#define WS_PTS  0                              // NPPAD f32
#define WS_JAUX 139264                         // 2048*5 f32 jets for aux points
#define WS_WT1  180224                         // 128x128 bf16, [out][in]
#define WS_WT2  212992                         // 128x128 bf16, [out][in]
#define WS_WT3  245760                         // 16x128 bf16 head panel (col 0 = W3)

// ---------------- branch-free fast tanh (native exp + rcp) ----------------
__device__ __forceinline__ float fast_tanh(float x) {
    float cx = fminf(fmaxf(x, -10.f), 10.f);       // tanh saturated anyway
    float e  = __expf(2.f * cx);                   // v_exp_f32 (TRANS pipe)
    return (e - 1.f) * __builtin_amdgcn_rcpf(e + 1.f);
}

// ---------------- tanh 4-jet (Faa di Bruno) ----------------
__device__ __forceinline__ void tanh_jet(const float a[5], float y[5]) {
    float t  = fast_tanh(a[0]);
    float t2 = t * t;
    float f1 = 1.f - t2;                 // f'
    float f2 = -2.f * t * f1;            // f''
    float f3 = f1 * (6.f * t2 - 2.f);    // f'''
    float f4 = 8.f * t * f1 * (2.f - 3.f * t2); // f''''
    float a1 = a[1], a2 = a[2], a3 = a[3], a4 = a[4];
    y[0] = t;
    y[1] = f1 * a1;
    y[2] = f1 * a2 + f2 * a1 * a1;
    y[3] = f1 * a3 + 3.f * f2 * a1 * a2 + f3 * a1 * a1 * a1;
    y[4] = f1 * a4 + f2 * (4.f * a1 * a3 + 3.f * a2 * a2)
         + 6.f * f3 * a1 * a1 * a2 + f4 * a1 * a1 * a1 * a1;
}

// ---------------- Gauss-Legendre nodes (f64 Newton) ----------------
__global__ void gl_nodes_kernel(float* __restrict__ pts) {
    int i = blockIdx.x * blockDim.x + threadIdx.x;
    if (i >= N_QUAD) return;
    const int n = N_QUAD;
    double x = cos(3.14159265358979323846 * (i + 0.75) / (n + 0.5));
    for (int it = 0; it < 8; ++it) {
        double p0 = 1.0, p1 = x;
        for (int j = 2; j <= n; ++j) {
            double pj = ((2.0 * j - 1.0) * x * p1 - (j - 1.0) * p0) / (double)j;
            p0 = p1; p1 = pj;
        }
        double dp = n * (x * p1 - p0) / (x * x - 1.0);
        x -= p1 / dp;
    }
    pts[N_PTS + i] = (float)(0.5 * x + 0.5);   // map [-1,1] -> [0,1]
}

// ---------------- pack points + bf16-transpose weights ----------------
__global__ void prep_kernel(const float* __restrict__ x,
                            const float* __restrict__ data_x,
                            const float* __restrict__ W1,
                            const float* __restrict__ W2,
                            const float* __restrict__ W3,
                            float* __restrict__ pts,
                            __bf16* __restrict__ Wt1,
                            __bf16* __restrict__ Wt2,
                            __bf16* __restrict__ Wt3) {
    int i = blockIdx.x * blockDim.x + threadIdx.x;
    if (i < N_PTS) pts[i] = x[i];
    else if (i >= N_PTS + N_QUAD && i < N_PTS + N_QUAD + N_DATA)
        pts[i] = data_x[i - (N_PTS + N_QUAD)];
    else if (i == N_PTS + N_QUAD + N_DATA)     pts[i] = 0.f;  // s = 0
    else if (i == N_PTS + N_QUAD + N_DATA + 1) pts[i] = 1.f;  // s = L/X_C
    else if (i >= NPTOT && i < NPPAD)          pts[i] = 0.f;  // pad
    int j = i - NPPAD;
    if (j >= 0 && j < 16384) {        // Wt1[c][k] = W1[k][c]
        int c = j >> 7, k = j & 127;
        Wt1[j] = (__bf16)W1[k * 128 + c];
    }
    j -= 16384;
    if (j >= 0 && j < 16384) {
        int c = j >> 7, k = j & 127;
        Wt2[j] = (__bf16)W2[k * 128 + c];
    }
    j -= 16384;
    if (j >= 0 && j < 2048) {         // head panel: col 0 = W3, cols 1..15 = 0
        int c = j >> 7, k = j & 127;
        Wt3[j] = (c == 0) ? (__bf16)W3[k] : (__bf16)0.f;
    }
}

// ---------------- main jet-MLP kernel: WMMA bf16 -> f32 ----------------
__global__ void __launch_bounds__(128)
pinn_jet_kernel(const float* __restrict__ pts,
                const float* __restrict__ W0, const float* __restrict__ b0,
                const __bf16* __restrict__ Wt1, const float* __restrict__ b1,
                const __bf16* __restrict__ Wt2, const float* __restrict__ b2,
                const __bf16* __restrict__ Wt3, const float* __restrict__ b3,
                float* __restrict__ out, float* __restrict__ jaux) {
    // double-buffered bf16 jet activations: [buf][jet][16 pts][HS feats]
    __shared__ __align__(16) __bf16 Hbuf[2][5][16 * HS];

    const int tid  = threadIdx.x;
    const int lane = tid & 31;
    const int wave = tid >> 5;
    const int hl   = (lane >> 4) & 1;   // lane half (ISA K-split)
    const int lc   = lane & 15;         // lane within half
    const int g0   = blockIdx.x * 16;   // first point of this tile

    // warm WGP cache for 2nd-layer weights while layer 0/1 execute
    __builtin_prefetch(Wt2 + tid * 128, 0, 1);   // global_prefetch_b8

    // ---- layer 0: z = s*W0 + b0, jets (z, W0, 0, 0, 0) -> tanh jet ----
    for (int e = tid; e < 16 * 128; e += 128) {
        int p = e >> 7, c = e & 127;
        float s = pts[g0 + p];
        float a[5] = { s * W0[c] + b0[c], W0[c], 0.f, 0.f, 0.f };
        float y[5]; tanh_jet(a, y);
        #pragma unroll
        for (int k = 0; k < 5; ++k) Hbuf[0][k][p * HS + c] = (__bf16)y[k];
    }
    __syncthreads();

    // ---- layers 1 & 2: 5 jets x [16x128]@[128x128] on the WMMA pipe ----
    #pragma unroll
    for (int layer = 0; layer < 2; ++layer) {
        const __bf16* __restrict__ Wt = layer ? Wt2 : Wt1;
        const float*  __restrict__ bb = layer ? b2  : b1;
        __bf16 (*Hin)[16 * HS]  = Hbuf[layer];
        __bf16 (*Hout)[16 * HS] = Hbuf[layer ^ 1];

        for (int ct = wave; ct < 8; ct += 4) {       // output column tiles
            const int col  = ct * 16 + lc;
            const float bias = bb[col];
            v8f acc[5];
            #pragma unroll
            for (int k = 0; k < 5; ++k) {
                v8f z;
                #pragma unroll
                for (int r = 0; r < 8; ++r) z[r] = (k == 0) ? bias : 0.f;
                acc[k] = z;
            }
            #pragma unroll
            for (int kb = 0; kb < 128; kb += 32) {
                // B operand: Wt[col][kb + 8*hl .. +7 | +16..+23]
                const __bf16* bp = Wt + col * 128 + kb + 8 * hl;
                v8bf blo = *(const v8bf*)bp;
                v8bf bhi = *(const v8bf*)(bp + 16);
                v16bf bm = __builtin_shufflevector(blo, bhi,
                    0,1,2,3,4,5,6,7,8,9,10,11,12,13,14,15);
                #pragma unroll
                for (int k = 0; k < 5; ++k) {
                    // A operand: Hin[k][row lc][kb + 8*hl .. +7 | +16..+23]
                    const __bf16* ap = &Hin[k][lc * HS + kb + 8 * hl];
                    v8bf alo = *(const v8bf*)ap;
                    v8bf ahi = *(const v8bf*)(ap + 16);
                    v16bf am = __builtin_shufflevector(alo, ahi,
                        0,1,2,3,4,5,6,7,8,9,10,11,12,13,14,15);
                    acc[k] = __builtin_amdgcn_wmma_f32_16x16x32_bf16(
                        false, am, false, bm, (short)0, acc[k], false, false);
                }
            }
            // pointwise tanh-jet on the 16x16 f32 tile (C/D layout: p = r + 8*hl)
            #pragma unroll
            for (int r = 0; r < 8; ++r) {
                int p = r + 8 * hl;
                float a[5] = { acc[0][r], acc[1][r], acc[2][r], acc[3][r], acc[4][r] };
                float y[5]; tanh_jet(a, y);
                #pragma unroll
                for (int k = 0; k < 5; ++k) Hout[k][p * HS + col] = (__bf16)y[k];
            }
        }
        __syncthreads();
    }

    // ---- head via WMMA: B panel has W3 in column 0, zeros elsewhere ----
    // waves split the 5 jets; trip counts are wave-uniform so EXEC==all-1s at WMMA
    for (int k = wave; k < 5; k += 4) {
        v8f acc;
        #pragma unroll
        for (int r = 0; r < 8; ++r) acc[r] = 0.f;
        #pragma unroll
        for (int kb = 0; kb < 128; kb += 32) {
            const __bf16* bp = Wt3 + lc * 128 + kb + 8 * hl;
            v8bf blo = *(const v8bf*)bp;
            v8bf bhi = *(const v8bf*)(bp + 16);
            v16bf bm = __builtin_shufflevector(blo, bhi,
                0,1,2,3,4,5,6,7,8,9,10,11,12,13,14,15);
            const __bf16* ap = &Hbuf[0][k][lc * HS + kb + 8 * hl];
            v8bf alo = *(const v8bf*)ap;
            v8bf ahi = *(const v8bf*)(ap + 16);
            v16bf am = __builtin_shufflevector(alo, ahi,
                0,1,2,3,4,5,6,7,8,9,10,11,12,13,14,15);
            acc = __builtin_amdgcn_wmma_f32_16x16x32_bf16(
                false, am, false, bm, (short)0, acc, false, false);
        }
        float bias3 = (k == 0) ? b3[0] : 0.f;
        if (lc == 0) {                       // column 0 of D = u_k[p]
            #pragma unroll
            for (int r = 0; r < 8; ++r) {
                int p = r + 8 * hl;
                int g = g0 + p;
                float u = acc[r] + bias3;
                if (g < N_PTS) {
                    // (w*W_C, dw*W_C, M=-ddw, Q=-dddw, ddddw*W_C)  [EI*W_C == X_C == 1]
                    const float scale[5] = { W_CONST, W_CONST, -1.f, -1.f, W_CONST };
                    out[k * N_PTS + g] = u * scale[k];
                } else {
                    int j = g - N_PTS;
                    if (j < 2048) jaux[j * 5 + k] = u;   // quad / data / boundary jets
                }
            }
        }
    }
}

// ---------------- loss reduction ----------------
__global__ void __launch_bounds__(256)
reduce_kernel(const float* __restrict__ jaux,
              const float* __restrict__ data_w,
              float* __restrict__ out) {
    __shared__ float s1[256], s2[256];
    int tid = threadIdx.x;
    float fsq = 0.f, dsum = 0.f;
    for (int i = tid; i < N_QUAD; i += 256) {       // sum (w'')^2 at GL nodes
        float d2 = jaux[i * 5 + 2];
        fsq += d2 * d2;
    }
    for (int i = tid; i < N_DATA; i += 256) {       // data MSE (pred - w/W_C)^2
        float pred = jaux[(N_QUAD + i) * 5 + 0];
        float d = pred - data_w[i] * (1.f / W_CONST);
        dsum += d * d;
    }
    s1[tid] = fsq; s2[tid] = dsum;
    __syncthreads();
    for (int o = 128; o > 0; o >>= 1) {
        if (tid < o) { s1[tid] += s1[tid + o]; s2[tid] += s2[tid + o]; }
        __syncthreads();
    }
    if (tid == 0) {
        float sum_fsq = s1[0];
        float res_data = s2[0] / (float)N_DATA;
        const float* j0 = jaux + (N_QUAD + N_DATA) * 5;  // s = 0
        const float* jL = j0 + 5;                        // s = 1
        float w0 = j0[0], dw0 = j0[1];
        float wL = jL[0], ddwL = jL[2], dddwL = jL[3];
        // quad = sum(wq)*sum(fsq) = 2*sum_fsq ; 0.5*EI*W_C^2*0.5 = 0.375
        float inter_en = 0.75f * sum_fsq;
        float exter_en = 5.f * W_CONST * wL;             // F_LOAD * W_C * wL
        float res_pel  = inter_en - exter_en;
        float res_dir  = 0.5f * (w0 * w0 + dw0 * dw0);
        float Fb = -dddwL;                                // EI*W_C/X_C^3 == 1
        float t1 = Fb * 0.2f - 1.f;
        float res_neu = 0.5f * (t1 * t1 + ddwL * ddwL);
        out[5 * N_PTS] = res_pel + 0.5f * (res_dir + res_neu) + res_data;
    }
}

// ---------------- host launcher ----------------
extern "C" void kernel_launch(void* const* d_in, const int* in_sizes, int n_in,
                              void* d_out, int out_size, void* d_ws, size_t ws_size,
                              hipStream_t stream) {
    const float* W0 = (const float*)d_in[0];
    const float* b0 = (const float*)d_in[1];
    const float* W1 = (const float*)d_in[2];
    const float* b1 = (const float*)d_in[3];
    const float* W2 = (const float*)d_in[4];
    const float* b2 = (const float*)d_in[5];
    const float* W3 = (const float*)d_in[6];
    const float* b3 = (const float*)d_in[7];
    const float* x  = (const float*)d_in[8];
    const float* dx = (const float*)d_in[9];
    const float* dw = (const float*)d_in[10];
    float* out = (float*)d_out;

    char* ws = (char*)d_ws;
    float*  pts  = (float*)(ws + WS_PTS);
    float*  jaux = (float*)(ws + WS_JAUX);
    __bf16* Wt1  = (__bf16*)(ws + WS_WT1);
    __bf16* Wt2  = (__bf16*)(ws + WS_WT2);
    __bf16* Wt3  = (__bf16*)(ws + WS_WT3);

    int prep_threads = NPPAD + 2 * 16384 + 2048;
    prep_kernel<<<(prep_threads + 255) / 256, 256, 0, stream>>>(
        x, dx, W1, W2, W3, pts, Wt1, Wt2, Wt3);
    gl_nodes_kernel<<<(N_QUAD + 127) / 128, 128, 0, stream>>>(pts);
    pinn_jet_kernel<<<NTILES, 128, 0, stream>>>(
        pts, W0, b0, Wt1, b1, Wt2, b2, Wt3, b3, out, jaux);
    reduce_kernel<<<1, 256, 0, stream>>>(jaux, dw, out);
}
